// VDSR_Rec_7284264534539
// MI455X (gfx1250) — matrix operand order repaired and checked by reading, not compiled, not run on hardware
//
#include <hip/hip_runtime.h>

typedef __attribute__((ext_vector_type(16))) __bf16 v16bf;
typedef __attribute__((ext_vector_type(8)))  float  v8f;

typedef uint4 uint4_a __attribute__((may_alias));

struct U8 { unsigned int x[8]; };

// fixed problem dims
#define DH 256
#define DW 256
#define DHW 65536

__device__ __forceinline__ unsigned short f2bf(float f) {
  unsigned int u = __float_as_uint(f);
  u += 0x7fffu + ((u >> 16) & 1u);      // round-to-nearest-even bf16
  return (unsigned short)(u >> 16);
}

template <int ACT>
__device__ __forceinline__ float act_apply(float v) {
  if (ACT == 1) return v > 0.f ? v : 0.f;            // relu
  if (ACT == 2) return v > 0.f ? v : 0.1f * v;       // lrelu(0.1)
  if (ACT == 3) return 1.f / (1.f + __expf(-v));     // sigmoid
  return v;
}

__device__ __forceinline__ v16bf mk_frag(uint4 a, uint4 b) {
  U8 u;
  u.x[0] = a.x; u.x[1] = a.y; u.x[2] = a.z; u.x[3] = a.w;
  u.x[4] = b.x; u.x[5] = b.y; u.x[6] = b.z; u.x[7] = b.w;
  return __builtin_bit_cast(v16bf, u);
}

// ---------------------------------------------------------------------------
// Implicit-GEMM conv (1x1 or 3x3, pad=k/2) via WMMA bf16, fully specialized.
// Block = 128 threads (4 waves), one NPIX-pixel tile (one row segment).
// im2col panel (NPIX x KPAD bf16, input-activation fused) staged once in LDS,
// k-major per pixel -> B fragment = two ds_load_b128. Weight rows KPAD-strided
// -> A fragment = two 16B global loads, reused across NPIX/16 B tiles.
// Wave wv covers out-channels [m0, m0+16); grid.y * 64 must cover Cout.
// ---------------------------------------------------------------------------
template <int KPAD, int NPIX, int KS, int AIN, int AOUT, bool RES>
__global__ __launch_bounds__(128) void conv_wmma(
    const float* __restrict__ xA, const float* __restrict__ xB, int csplit,
    const unsigned short* __restrict__ wbf, const float* __restrict__ bias,
    const float* __restrict__ resA, const float* __restrict__ resB, int rsplit,
    float* __restrict__ y, int Cin, int Cout)
{
  constexpr int KL   = KPAD + 8;     // LDS row stride (shorts), keeps 16B align
  constexpr int NP16 = NPIX / 16;
  __shared__ unsigned short sB[NPIX * KL];

  const int Ktot = (KS == 3) ? Cin * 9 : Cin;
  const int p0  = blockIdx.x * NPIX;           // pixel tile start (same row)
  const int b   = p0 >> 16;                    // / DHW
  const int hw0 = p0 & (DHW - 1);
  const int h0  = hw0 >> 8;                    // / DW
  const int w0  = hw0 & (DW - 1);

  // ---- stage im2col panel: sB[n*KL + k] = bf16(act_in(x[k -> ci,dy,dx]))
  for (int idx = threadIdx.x; idx < NPIX * KPAD; idx += 128) {
    int k = idx / NPIX;          // consecutive tid -> consecutive n (coalesced)
    int n = idx & (NPIX - 1);
    float v = 0.f;
    if (k < Ktot) {
      int ci, hh, ww;
      if (KS == 3) { ci = k / 9; int r = k % 9; hh = h0 + r / 3 - 1; ww = w0 + n + r % 3 - 1; }
      else         { ci = k; hh = h0; ww = w0 + n; }
      if ((unsigned)hh < (unsigned)DH && (unsigned)ww < (unsigned)DW) {
        float vv;
        if (ci < csplit)
          vv = xA[(b * csplit + ci) * DHW + hh * DW + ww];
        else
          vv = xB[(b * (Cin - csplit) + (ci - csplit)) * DHW + hh * DW + ww];
        v = act_apply<AIN>(vv);
      }
    }
    sB[n * KL + k] = f2bf(v);
  }
  __syncthreads();

  // ---- per-wave 16 x NPIX output tile
  const int lane = threadIdx.x & 31;
  const int wv   = threadIdx.x >> 5;
  const int m0   = blockIdx.y * 64 + wv * 16;
  const int nB   = lane & 15;
  const int kHB  = (lane >= 16) ? 8 : 0;           // B-frag K base
  const int mA   = m0 + (lane & 15);               // A-frag M row
  const int kHA  = (lane >= 16) ? 8 : 0;           // A-frag K base

  const unsigned short* wrow = wbf + (size_t)mA * KPAD;

  v8f acc[NP16];
#pragma unroll
  for (int t = 0; t < NP16; ++t) acc[t] = (v8f){0.f,0.f,0.f,0.f,0.f,0.f,0.f,0.f};

  for (int kc = 0; kc < KPAD; kc += 32) {
    const uint4_a* pa = (const uint4_a*)(wrow + kc + kHA);
    v16bf af = mk_frag(pa[0], pa[2]);    // K = kHA+[0,8), kHA+[16,24)
#pragma unroll
    for (int t = 0; t < NP16; ++t) {
      const uint4_a* pb = (const uint4_a*)(sB + (t * 16 + nB) * KL + kc + kHB);
      v16bf bf = mk_frag(pb[0], pb[2]);  // K = kHB+[0,8), kHB+[16,24)
      acc[t] = __builtin_amdgcn_wmma_f32_16x16x32_bf16(false, af, false, bf, (short)0,
                                                       acc[t], false, false);
    }
  }

  const int mhi = (lane >> 4) * 8;
#pragma unroll
  for (int t = 0; t < NP16; ++t) {
    const int w = w0 + t * 16 + nB;
#pragma unroll
    for (int r = 0; r < 8; ++r) {
      int m = m0 + r + mhi;
      float v = acc[t][r] + bias[m];
      if constexpr (RES) {
        float rv;
        if (m < rsplit)
          rv = resA[(b * rsplit + m) * DHW + h0 * DW + w];
        else
          rv = resB[(b * (Cout - rsplit) + (m - rsplit)) * DHW + h0 * DW + w];
        v += rv;
      }
      v = act_apply<AOUT>(v);
      y[(b * Cout + m) * DHW + h0 * DW + w] = v;
    }
  }
}

// ---------------------------------------------------------------------------
// Fused KGenerator 1x1 (64 -> 576) + unfold3 apply + rgb2 combine.
// Block = 128 threads = 4 waves, one 16-pixel tile. 36 M-tiles of 16; wave w
// handles tiles [9w, 9w+9). lrelu(kx) staged n-major in LDS (stride 72) so
// B fragments are two ds_load_b128. Filter values staged in LDS; the 9-tap
// reduction is done deterministically per (c, pixel).
// ---------------------------------------------------------------------------
__global__ __launch_bounds__(128) void kgen_apply(
    const float* __restrict__ kx,           // (Bn,64,H,W) pre-lrelu
    const unsigned short* __restrict__ kgw, // bf16 (576,64)
    const float* __restrict__ kgb,          // (576)
    const float* __restrict__ rgb,          // (Bn,64,H,W)
    const float* __restrict__ att,          // (Bn,64,H,W)
    const float* __restrict__ beta1,
    float* __restrict__ rgb2)
{
  __shared__ unsigned short albf[16 * 72];  // lrelu(kx) bf16, [n][k], stride 72
  __shared__ float filt[576 * 16];          // dynamic filters, [c9][n]

  const int tid = threadIdx.x;
  const int p0  = blockIdx.x * 16;
  const int b   = p0 >> 16;
  const int hw0 = p0 & (DHW - 1);
  const int h0  = hw0 >> 8;
  const int w0  = hw0 & (DW - 1);

  for (int idx = tid; idx < 1024; idx += 128) {
    int k = idx >> 4, n = idx & 15;
    float v = kx[(b * 64 + k) * DHW + hw0 + n];
    albf[n * 72 + k] = f2bf(v > 0.f ? v : 0.1f * v);
  }
  __syncthreads();

  const int wv = tid >> 5, lane = tid & 31;
  const int nB  = lane & 15;
  const int kHB = (lane >= 16) ? 8 : 0;
  const int kHA = (lane >= 16) ? 8 : 0;

  const unsigned short* brow = albf + nB * 72;
  const uint4_a* pb0 = (const uint4_a*)(brow + kHB);        // kc = 0
  const uint4_a* pb1 = (const uint4_a*)(brow + 32 + kHB);   // kc = 32
  v16bf bf0 = mk_frag(pb0[0], pb0[2]);
  v16bf bf1 = mk_frag(pb1[0], pb1[2]);

  const int mhi = (lane >> 4) * 8;
  for (int t = 0; t < 9; ++t) {
    int m0 = (wv * 9 + t) * 16;
    const unsigned short* wrow = kgw + (m0 + (lane & 15)) * 64;
    const uint4_a* pa = (const uint4_a*)(wrow + kHA);
    v16bf af0 = mk_frag(pa[0], pa[2]);
    v16bf af1 = mk_frag(pa[4], pa[6]);
    v8f acc = {0.f,0.f,0.f,0.f,0.f,0.f,0.f,0.f};
    acc = __builtin_amdgcn_wmma_f32_16x16x32_bf16(false, af0, false, bf0, (short)0, acc, false, false);
    acc = __builtin_amdgcn_wmma_f32_16x16x32_bf16(false, af1, false, bf1, (short)0, acc, false, false);
#pragma unroll
    for (int r = 0; r < 8; ++r) {
      int m = m0 + r + mhi;                 // c9 in [0,576)
      filt[m * 16 + nB] = acc[r] + kgb[m];
    }
  }
  __syncthreads();

  const float beta = beta1[0];
  for (int idx = tid; idx < 1024; idx += 128) {
    int c = idx >> 4, n = idx & 15;
    int ww = w0 + n;
    float s = 0.f;
#pragma unroll
    for (int kk = 0; kk < 9; ++kk) {
      int yy = h0 + kk / 3 - 1, xx = ww + kk % 3 - 1;
      float xv = 0.f;
      if ((unsigned)yy < (unsigned)DH && (unsigned)xx < (unsigned)DW)
        xv = rgb[(b * 64 + c) * DHW + yy * DW + xx];
      s += filt[(c * 9 + kk) * 16 + n] * xv;
    }
    int off = (b * 64 + c) * DHW + hw0 + n;
    rgb2[off] = att[off] * rgb[off] + beta * s;
  }
}

// ---------------------------------------------------------------------------
// Elementwise / reduction helpers (dims hardcoded, exact grids)
// ---------------------------------------------------------------------------
// 256->128 bilinear (align_corners=False, scale 2) == 2x2 average
__global__ void down_k(const float* __restrict__ x, float* __restrict__ y)
{
  int i = blockIdx.x * 256 + threadIdx.x;      // total = Bn*C*128*128
  int wo = i & 127;
  int ho = (i >> 7) & 127;
  int bc = i >> 14;
  const float* p = x + bc * DHW + (ho * 2) * DW + wo * 2;
  y[i] = 0.25f * (p[0] + p[1] + p[DW] + p[DW + 1]);
}

// 128->256 bilinear upsample + |dep - up|
__global__ void up_absdiff_k(const float* __restrict__ dn, const float* __restrict__ dep,
                             float* __restrict__ out)
{
  int i = blockIdx.x * 256 + threadIdx.x;      // total = Bn*C*256*256
  int wo = i & 255;
  int ho = (i >> 8) & 255;
  int bc = i >> 16;
  float sy = fminf(fmaxf(ho * 0.5f - 0.25f, 0.f), 127.f);
  int y0 = (int)sy; int y1 = min(y0 + 1, 127); float fy = sy - y0;
  float sx = fminf(fmaxf(wo * 0.5f - 0.25f, 0.f), 127.f);
  int x0 = (int)sx; int x1 = min(x0 + 1, 127); float fx = sx - x0;
  const float* base = dn + bc * (128 * 128);
  float v00 = base[y0 * 128 + x0], v01 = base[y0 * 128 + x1];
  float v10 = base[y1 * 128 + x0], v11 = base[y1 * 128 + x1];
  float v = (v00 * (1.f - fx) + v01 * fx) * (1.f - fy) + (v10 * (1.f - fx) + v11 * fx) * fy;
  out[i] = fabsf(dep[i] - v);
}

__global__ void chan_mean_k(const float* __restrict__ x, float* __restrict__ out)
{
  __shared__ float sm[256];
  const float* p = x + blockIdx.x * DHW;
  float s = 0.f;
  for (int i = threadIdx.x; i < DHW; i += 256) s += p[i];
  sm[threadIdx.x] = s;
  __syncthreads();
  for (int st = 128; st > 0; st >>= 1) {
    if (threadIdx.x < st) sm[threadIdx.x] += sm[threadIdx.x + st];
    __syncthreads();
  }
  if (threadIdx.x == 0) out[blockIdx.x] = sm[0] * (1.f / (float)DHW);
}

template <int C, int Cr>
__global__ void ca_fc_k(const float* __restrict__ mean,
                        const float* __restrict__ wd, const float* __restrict__ bd,
                        const float* __restrict__ wu, const float* __restrict__ bu,
                        float* __restrict__ svec)
{
  __shared__ float hid[Cr];
  __shared__ float mloc[C];
  int b = blockIdx.x;
  mloc[threadIdx.x] = mean[b * C + threadIdx.x];
  __syncthreads();
  if ((int)threadIdx.x < Cr) {
    float s = bd[threadIdx.x];
    for (int j = 0; j < C; ++j) s += wd[threadIdx.x * C + j] * mloc[j];
    hid[threadIdx.x] = s > 0.f ? s : 0.f;
  }
  __syncthreads();
  {
    float s = bu[threadIdx.x];
    for (int j = 0; j < Cr; ++j) s += wu[threadIdx.x * Cr + j] * hid[j];
    svec[b * C + threadIdx.x] = 1.f / (1.f + __expf(-s));
  }
}

// y = cat(xA,xB)[i] + res[i] * s[b,c]; C power of two; exact grid
template <int C, int CS>
__global__ void ca_combine_k(const float* __restrict__ xA, const float* __restrict__ xB,
                             const float* __restrict__ res, const float* __restrict__ svec,
                             float* __restrict__ y)
{
  int i = blockIdx.x * 256 + threadIdx.x;
  int hw = i & (DHW - 1);
  int t = i >> 16;                 // b*C + c
  int c = t & (C - 1);
  int b = t >> (C == 64 ? 6 : 7);
  float xv;
  if (c < CS) xv = xA[(b * CS + c) * DHW + hw];
  else        xv = xB[(b * (C - CS) + (c - CS)) * DHW + hw];
  y[i] = xv + res[i] * svec[b * C + c];
}

__global__ void diffstats_k(const float* __restrict__ diff, float* __restrict__ amap)
{
  int i = blockIdx.x * 256 + threadIdx.x;      // total = Bn*HW
  int b = i >> 16, hw = i & (DHW - 1);
  const float* p = diff + b * 64 * DHW + hw;
  float s = 0.f, mx = -3.4e38f;
  for (int c = 0; c < 64; ++c) { float v = p[c * DHW]; s += v; mx = fmaxf(mx, v); }
  amap[(b * 2 + 0) * DHW + hw] = s * (1.f / 64.f);
  amap[(b * 2 + 1) * DHW + hw] = mx;
}

__global__ void cvt_bf16_k(const float* __restrict__ x, unsigned short* __restrict__ y, int n)
{
  int i = blockIdx.x * 256 + threadIdx.x;
  if (i < n) y[i] = f2bf(x[i]);
}

// convert with row padding: y is (rows, kout), x is (rows, kin), pad with 0
__global__ void cvt_pad_k(const float* __restrict__ x, unsigned short* __restrict__ y,
                          int rows, int kin, int kout)
{
  int i = blockIdx.x * 256 + threadIdx.x;
  if (i >= rows * kout) return;
  int r = i / kout, k = i % kout;
  y[i] = (k < kin) ? f2bf(x[r * kin + k]) : (unsigned short)0;
}

// ---------------------------------------------------------------------------
extern "C" void kernel_launch(void* const* d_in, const int* in_sizes, int n_in,
                              void* d_out, int out_size, void* d_ws, size_t ws_size,
                              hipStream_t stream)
{
  (void)in_sizes; (void)n_in; (void)out_size; (void)ws_size;
  const int Bn = 2, C = 64;
  const size_t N1 = (size_t)Bn * C * DHW;      // 8,388,608

  const float* dep = (const float*)d_in[0];
  const float* rgb = (const float*)d_in[1];
  const float* cd_w      = (const float*)d_in[2];
  const float* cd_b      = (const float*)d_in[3];
  const float* du_w      = (const float*)d_in[4];
  const float* du_b      = (const float*)d_in[5];
  const float* kg_c1_w   = (const float*)d_in[6];
  const float* kg_c1_b   = (const float*)d_in[7];
  const float* kg_c2_w   = (const float*)d_in[8];
  const float* kg_c2_b   = (const float*)d_in[9];
  const float* kg_cad_w  = (const float*)d_in[10];
  const float* kg_cad_b  = (const float*)d_in[11];
  const float* kg_cau_w  = (const float*)d_in[12];
  const float* kg_cau_b  = (const float*)d_in[13];
  const float* kg_tail_w = (const float*)d_in[14];
  const float* kg_tail_b = (const float*)d_in[15];
  const float* kg_w      = (const float*)d_in[16];
  const float* kg_b      = (const float*)d_in[17];
  const float* rg_c1_w   = (const float*)d_in[18];
  const float* rg_c1_b   = (const float*)d_in[19];
  const float* rg_c2_w   = (const float*)d_in[20];
  const float* rg_c2_b   = (const float*)d_in[21];
  const float* rg_cad_w  = (const float*)d_in[22];
  const float* rg_cad_b  = (const float*)d_in[23];
  const float* rg_cau_w  = (const float*)d_in[24];
  const float* rg_cau_b  = (const float*)d_in[25];
  const float* rg_tail_w = (const float*)d_in[26];
  const float* rg_tail_b = (const float*)d_in[27];
  const float* beta      = (const float*)d_in[28];
  const float* re_w      = (const float*)d_in[29];
  const float* re_b      = (const float*)d_in[30];

  float* out_re   = (float*)d_out;
  float* out_att  = out_re + N1;
  float* out_diff = out_att + N1;

  float* ws   = (float*)d_ws;
  float* t0   = ws;              // 64ch buffer
  float* t1   = ws + N1;         // 64ch buffer
  float* t2   = ws + 2 * N1;     // 64ch buffer
  float* u0   = t1;              // 128ch buffer overlays t1+t2 (after kx consumed)
  float* u1   = ws + 3 * N1;     // 128ch buffer (2*N1)
  float* down = ws + 5 * N1;     // N1/4
  float* amap = down + N1 / 4;   // Bn*2*HW
  float* means = amap + (size_t)Bn * 2 * DHW;
  float* svec  = means + 256;
  unsigned short* wb = (unsigned short*)(svec + 256);
  unsigned short* bw_cd      = wb; wb += 36864;    // 64 x 576
  unsigned short* bw_kg_c1   = wb; wb += 36864;
  unsigned short* bw_kg_c2   = wb; wb += 36864;
  unsigned short* bw_kg_tail = wb; wb += 36864;
  unsigned short* bw_kg      = wb; wb += 36864;    // 576 x 64
  unsigned short* bw_du      = wb; wb += 64 * 32;  // 64 x 32 (K=18 padded)
  unsigned short* bw_rg_c1   = wb; wb += 147456;   // 128 x 1152
  unsigned short* bw_rg_c2   = wb; wb += 147456;
  unsigned short* bw_rg_tail = wb; wb += 147456;
  unsigned short* bw_re      = wb; wb += 8192;     // 64 x 128

  // weight fp32 -> bf16
  auto cvt = [&](const float* s, unsigned short* d, int n) {
    cvt_bf16_k<<<(n + 255) / 256, 256, 0, stream>>>(s, d, n);
  };
  cvt(cd_w, bw_cd, 36864);
  cvt(kg_c1_w, bw_kg_c1, 36864);
  cvt(kg_c2_w, bw_kg_c2, 36864);
  cvt(kg_tail_w, bw_kg_tail, 36864);
  cvt(kg_w, bw_kg, 36864);
  cvt_pad_k<<<(64 * 32 + 255) / 256, 256, 0, stream>>>(du_w, bw_du, 64, 18, 32);
  cvt(rg_c1_w, bw_rg_c1, 147456);
  cvt(rg_c2_w, bw_rg_c2, 147456);
  cvt(rg_tail_w, bw_rg_tail, 147456);
  cvt(re_w, bw_re, 8192);

  const int tiles32 = (int)((size_t)Bn * DHW / 32);   // 4096
  const int tiles16 = (int)((size_t)Bn * DHW / 16);   // 8192
  dim3 blk(128);

  // 1) downscale dep 256->128 (2x2 average)
  down_k<<<(Bn * C * 128 * 128) / 256, 256, 0, stream>>>(dep, down);
  // 2) diff_intra = |dep - up(down)|  -> d_out section 3
  up_absdiff_k<<<(unsigned)(N1 / 256), 256, 0, stream>>>(down, dep, out_diff);
  // 3) t0 = conv_diff(lrelu(diff))
  conv_wmma<576, 32, 3, 2, 0, false><<<dim3(tiles32, 1), blk, 0, stream>>>(
      out_diff, nullptr, C, bw_cd, cd_b, nullptr, nullptr, C, t0, C, C);
  // 4) KG rcab: t1 = relu(c1(t0)); t2 = c2(t1)
  conv_wmma<576, 32, 3, 0, 1, false><<<dim3(tiles32, 1), blk, 0, stream>>>(
      t0, nullptr, C, bw_kg_c1, kg_c1_b, nullptr, nullptr, C, t1, C, C);
  conv_wmma<576, 32, 3, 0, 0, false><<<dim3(tiles32, 1), blk, 0, stream>>>(
      t1, nullptr, C, bw_kg_c2, kg_c2_b, nullptr, nullptr, C, t2, C, C);
  // channel attention
  chan_mean_k<<<Bn * C, 256, 0, stream>>>(t2, means);
  ca_fc_k<64, 4><<<Bn, 64, 0, stream>>>(means, kg_cad_w, kg_cad_b, kg_cau_w, kg_cau_b, svec);
  ca_combine_k<64, 64><<<(unsigned)(N1 / 256), 256, 0, stream>>>(t0, nullptr, t2, svec, t1);
  // KG tail + residual: t2 = tail(t1) + t0   (== kx)
  conv_wmma<576, 32, 3, 0, 0, true><<<dim3(tiles32, 1), blk, 0, stream>>>(
      t1, nullptr, C, bw_kg_tail, kg_tail_b, t0, nullptr, C, t2, C, C);
  // 5) attention map: amap = [mean_c, max_c](diff); att = sigmoid(conv_du(amap))
  diffstats_k<<<(Bn * DHW) / 256, 256, 0, stream>>>(out_diff, amap);
  conv_wmma<32, 32, 3, 0, 3, false><<<dim3(tiles32, 1), blk, 0, stream>>>(
      amap, nullptr, 2, bw_du, du_b, nullptr, nullptr, C, out_att, 2, C);
  // 6) fused kg_conv + unfold apply + rgb2 = att*rgb + beta*out_rgb -> t0
  kgen_apply<<<tiles16, blk, 0, stream>>>(t2, bw_kg, kg_b, rgb, out_att, beta, t0);
  // 7) rg group on cat(dep, rgb2)  (128 ch)
  conv_wmma<1152, 16, 3, 0, 1, false><<<dim3(tiles16, 2), blk, 0, stream>>>(
      dep, t0, 64, bw_rg_c1, rg_c1_b, nullptr, nullptr, 128, u0, 128, 128);
  conv_wmma<1152, 16, 3, 0, 0, false><<<dim3(tiles16, 2), blk, 0, stream>>>(
      u0, nullptr, 128, bw_rg_c2, rg_c2_b, nullptr, nullptr, 128, u1, 128, 128);
  chan_mean_k<<<Bn * 128, 256, 0, stream>>>(u1, means);
  ca_fc_k<128, 8><<<Bn, 128, 0, stream>>>(means, rg_cad_w, rg_cad_b, rg_cau_w, rg_cau_b, svec);
  ca_combine_k<128, 64><<<(unsigned)(2 * N1 / 256), 256, 0, stream>>>(dep, t0, u1, svec, u0);
  conv_wmma<1152, 16, 3, 0, 0, true><<<dim3(tiles16, 2), blk, 0, stream>>>(
      u0, nullptr, 128, bw_rg_tail, rg_tail_b, dep, t0, 64, u1, 128, 128);
  // 8) out_re = conv_re(u1)  (1x1, 128 -> 64)
  conv_wmma<128, 32, 1, 0, 0, false><<<dim3(tiles32, 1), blk, 0, stream>>>(
      u1, nullptr, 128, bw_re, re_b, nullptr, nullptr, C, out_re, 128, C);
}
